// Field_Gradient_Optimizer_20409684591249
// MI455X (gfx1250) — matrix operand order repaired and checked by reading, not compiled, not run on hardware
//
#include <hip/hip_runtime.h>
#include <hip/hip_bf16.h>

typedef __attribute__((ext_vector_type(16))) _Float16 v16h;
typedef __attribute__((ext_vector_type(8)))  float    v8f;

#define OMEGA 30.0f
#define H 128
#define NVAC 16

__device__ __forceinline__ float nn0(float x) { return __builtin_isfinite(x) ? x : 0.0f; }
__device__ __forceinline__ float clamp25(float x) { return fminf(25.0f, fmaxf(-25.0f, x)); }

// ---------------------------------------------------------------------------
// Reductions for center = mean(pos) and radius = max(||pos-center||)
// ---------------------------------------------------------------------------
__global__ void sum_pos_kernel(const float* __restrict__ pos, int N, float* __restrict__ out) {
    __shared__ float sx[256], sy[256], sz[256];
    int tid = threadIdx.x;
    float ax = 0.f, ay = 0.f, az = 0.f;
    for (int i = blockIdx.x * 256 + tid; i < N; i += 256 * gridDim.x) {
        ax += pos[i * 3 + 0]; ay += pos[i * 3 + 1]; az += pos[i * 3 + 2];
    }
    sx[tid] = ax; sy[tid] = ay; sz[tid] = az;
    __syncthreads();
    for (int s = 128; s > 0; s >>= 1) {
        if (tid < s) { sx[tid] += sx[tid + s]; sy[tid] += sy[tid + s]; sz[tid] += sz[tid + s]; }
        __syncthreads();
    }
    if (tid == 0) {
        out[blockIdx.x * 3 + 0] = sx[0];
        out[blockIdx.x * 3 + 1] = sy[0];
        out[blockIdx.x * 3 + 2] = sz[0];
    }
}

__global__ void finish_center_kernel(const float* __restrict__ sums, int nblk, int N,
                                     float* __restrict__ center) {
    if (threadIdx.x == 0 && blockIdx.x == 0) {
        float x = 0.f, y = 0.f, z = 0.f;
        for (int i = 0; i < nblk; ++i) { x += sums[i * 3]; y += sums[i * 3 + 1]; z += sums[i * 3 + 2]; }
        float inv = 1.0f / (float)N;
        center[0] = x * inv; center[1] = y * inv; center[2] = z * inv;
    }
}

__global__ void max_rad_kernel(const float* __restrict__ pos, int N,
                               const float* __restrict__ center, float* __restrict__ outMax) {
    __shared__ float sm[256];
    int tid = threadIdx.x;
    float c0 = center[0], c1 = center[1], c2 = center[2];
    float mx = 0.0f;
    for (int i = blockIdx.x * 256 + tid; i < N; i += 256 * gridDim.x) {
        float dx = pos[i * 3] - c0, dy = pos[i * 3 + 1] - c1, dz = pos[i * 3 + 2] - c2;
        mx = fmaxf(mx, dx * dx + dy * dy + dz * dz);
    }
    sm[tid] = mx;
    __syncthreads();
    for (int s = 128; s > 0; s >>= 1) {
        if (tid < s) sm[tid] = fmaxf(sm[tid], sm[tid + s]);
        __syncthreads();
    }
    if (tid == 0) outMax[blockIdx.x] = sm[0];
}

__global__ void finish_rad_kernel(const float* __restrict__ maxs, int nblk, float* __restrict__ radius) {
    if (threadIdx.x == 0 && blockIdx.x == 0) {
        float m = 0.f;
        for (int i = 0; i < nblk; ++i) m = fmaxf(m, maxs[i]);
        radius[0] = fmaxf(sqrtf(m), 2.0f) * 1.35f;
    }
}

// ---------------------------------------------------------------------------
// Pre-swizzle W1 (f32 [128,128]) into f16 WMMA B-fragment layouts.
// Forward:  B[k][n] = W1[k][n]   (h0 @ W1)
// Backward: B[k][n] = W1[n][k]   (g1 @ W1^T)
// Fragment element (tile kb,nb; lane L; elem e):
//   k = kb*32 + (L>>4)*8 + (e<8 ? e : e+8),  n = nb*16 + (L&15)
// stored at [((kb*8+nb)*32 + L)*16 + e]
// ---------------------------------------------------------------------------
__global__ void prep_w1_kernel(const float* __restrict__ W1,
                               _Float16* __restrict__ w1f, _Float16* __restrict__ w1b) {
    int idx = blockIdx.x * blockDim.x + threadIdx.x;
    if (idx >= 16384) return;
    int e    = idx & 15;
    int lane = (idx >> 4) & 31;
    int t    = idx >> 9;        // 0..31 = kb*8 + nb
    int nb = t & 7, kb = t >> 3;
    int k = kb * 32 + (lane >> 4) * 8 + (e < 8 ? e : e + 8);
    int n = nb * 16 + (lane & 15);
    w1f[idx] = (_Float16)W1[k * H + n];
    w1b[idx] = (_Float16)W1[n * H + k];
}

// ---------------------------------------------------------------------------
// Main kernel: per wave32, 16 points. Forward h1 pre-activations and backward
// dL/dh0 via v_wmma_f32_16x16x32_f16; analytic SIREN gradient out + Huber.
// ---------------------------------------------------------------------------
__global__ void __launch_bounds__(128) field_grad_kernel(
    const float* __restrict__ pos, const float* __restrict__ forces,
    const float* __restrict__ W0, const float* __restrict__ b0,
    const float* __restrict__ b1, const float* __restrict__ W2,
    const _Float16* __restrict__ w1f, const _Float16* __restrict__ w1b,
    const float* __restrict__ center, float* __restrict__ gradsOut,
    float* __restrict__ hubPartials, int N)
{
    __shared__ float    uBuf[4][16][H];     // omega*cos(pre0) per [wave][point][feature]
    __shared__ _Float16 g1Buf[4][16][H];    // W2[j]*omega*cos(pre1) per [wave][point][feature]
    __shared__ float    hubParts[8];

    const int tid  = threadIdx.x;
    const int w    = tid >> 5;
    const int lane = tid & 31;
    const int hf   = lane >> 4;
    const int m    = lane & 15;
    const int nloc = lane & 15;
    const int base = (blockIdx.x * 4 + w) * 16;

    // ---- phase 0: perturbed coords for this tile's 16 points ----
    float cx = 0.f, cy = 0.f, cz = 0.f;
    if (lane < 16) {
        int p = base + lane; if (p >= N) p = N - 1;
        float px = pos[p * 3 + 0], py = pos[p * 3 + 1], pz = pos[p * 3 + 2];
        float fx = forces[p * 3 + 0], fy = forces[p * 3 + 1], fz = forces[p * 3 + 2];
        float dx = fx + 0.1f * (px - center[0]);
        float dy = fy + 0.1f * (py - center[1]);
        float dz = fz + 0.1f * (pz - center[2]);
        float nrm = fmaxf(sqrtf(dx * dx + dy * dy + dz * dz), 1e-8f);
        float inv = 0.01f / nrm;
        cx = px + dx * inv; cy = py + dy * inv; cz = pz + dz * inv;
    }
    float xm = __shfl(cx, m), ym = __shfl(cy, m), zm = __shfl(cz, m);

    // ---- phase 1: h0 directly in WMMA A-fragment layout; u -> LDS ----
    union AFrag { v16h v; _Float16 h[16]; };
    AFrag a[4];
#pragma unroll
    for (int kb = 0; kb < 4; ++kb) {
#pragma unroll
        for (int e = 0; e < 16; ++e) {
            int K = kb * 32 + hf * 8 + (e < 8 ? e : e + 8);   // 16-bit A layout K mapping
            float z0 = OMEGA * (W0[K] * xm + W0[H + K] * ym + W0[2 * H + K] * zm + b0[K]);
            a[kb].h[e] = (_Float16)__sinf(z0);
            uBuf[w][m][K] = OMEGA * __cosf(z0);
        }
    }

    // ---- forward GEMM: pre1 = omega*(h0 @ W1 + b1); store G1 = W2*omega*cos(pre1) ----
#pragma unroll 1
    for (int nb = 0; nb < 8; ++nb) {
        v8f c = {};
#pragma unroll
        for (int kb = 0; kb < 4; ++kb) {
            union { v16h v; uint4 q[2]; } bf;
            const uint4* src = (const uint4*)(w1f + ((size_t)((kb * 8 + nb) * 32 + lane)) * 16);
            bf.q[0] = src[0]; bf.q[1] = src[1];
            c = __builtin_amdgcn_wmma_f32_16x16x32_f16(false, a[kb].v, false, bf.v,
                                                       (short)0, c, false, false);
        }
        int f = nb * 16 + nloc;
        float w2f = W2[f];
        float b1f = b1[f];
#pragma unroll
        for (int v = 0; v < 8; ++v) {
            int p = v + 8 * hf;                 // C/D layout: M = v + 8*(lane>=16)
            float pre1 = OMEGA * (c[v] + b1f);
            float t = OMEGA * __cosf(pre1);
            g1Buf[w][p][f] = (_Float16)(w2f * t);
        }
    }
    __syncthreads();

    // ---- backward A fragments from LDS (two 8-element chunks per k block, table k order) ----
    AFrag a2[4];
#pragma unroll
    for (int kb = 0; kb < 4; ++kb) {
        union { v16h v; uint4 q[2]; } tmp;
        tmp.q[0] = *(const uint4*)&g1Buf[w][m][kb * 32 + hf * 8];
        tmp.q[1] = *(const uint4*)&g1Buf[w][m][kb * 32 + 16 + hf * 8];
        a2[kb].v = tmp.v;
    }

    // ---- backward GEMM: dh0 = G1 @ W1^T; grad = (dh0 * u) @ W0^T ----
    float gx[8] = {}, gy[8] = {}, gz[8] = {};
#pragma unroll 1
    for (int nb = 0; nb < 8; ++nb) {
        v8f c = {};
#pragma unroll
        for (int kb = 0; kb < 4; ++kb) {
            union { v16h v; uint4 q[2]; } bf;
            const uint4* src = (const uint4*)(w1b + ((size_t)((kb * 8 + nb) * 32 + lane)) * 16);
            bf.q[0] = src[0]; bf.q[1] = src[1];
            c = __builtin_amdgcn_wmma_f32_16x16x32_f16(false, a2[kb].v, false, bf.v,
                                                       (short)0, c, false, false);
        }
        int i = nb * 16 + nloc;
        float w00 = W0[i], w01 = W0[H + i], w02 = W0[2 * H + i];
#pragma unroll
        for (int v = 0; v < 8; ++v) {
            int p = v + 8 * hf;
            float g0 = c[v] * uBuf[w][p][i];
            gx[v] += g0 * w00; gy[v] += g0 * w01; gz[v] += g0 * w02;
        }
    }

    // reduce across the 16 lanes of each half-group
#pragma unroll
    for (int msk = 1; msk < 16; msk <<= 1) {
#pragma unroll
        for (int v = 0; v < 8; ++v) {
            gx[v] += __shfl_xor(gx[v], msk);
            gy[v] += __shfl_xor(gy[v], msk);
            gz[v] += __shfl_xor(gz[v], msk);
        }
    }

    if (nloc == 0) {                      // lanes 0 and 16 write 8 points each
        float hub = 0.0f;
#pragma unroll
        for (int v = 0; v < 8; ++v) {
            int p = base + v + 8 * hf;
            if (p < N) {
                float g[3] = { gx[v], gy[v], gz[v] };
#pragma unroll
                for (int k = 0; k < 3; ++k) {
                    float gc = clamp25(nn0(g[k]));
                    float tg = clamp25(nn0(forces[p * 3 + k]));
                    float d = gc - tg;
                    float ad = fabsf(d);
                    hub += (ad < 1.0f) ? 0.5f * d * d : ad - 0.5f;
                    gradsOut[p * 3 + k] = gc;
                }
            }
        }
        hubParts[w * 2 + hf] = hub;
    }
    __syncthreads();
    if (tid == 0) {
        float s = 0.f;
        for (int q = 0; q < 8; ++q) s += hubParts[q];
        hubPartials[blockIdx.x] = s;
    }
}

// ---------------------------------------------------------------------------
// Vacuum points: psi, clamped grad, gated analytic Hessian diagonal, spectral.
// One wave per point; lanes split the 128 hidden units.
// ---------------------------------------------------------------------------
__global__ void __launch_bounds__(512) vacuum_kernel(
    const float* __restrict__ W0, const float* __restrict__ b0,
    const float* __restrict__ W1, const float* __restrict__ b1,
    const float* __restrict__ W2, const float* __restrict__ b2,
    const float* __restrict__ center, const float* __restrict__ radius,
    float* __restrict__ psiOut, float* __restrict__ gvOut, float* __restrict__ specOut)
{
    __shared__ float h0s[NVAC][H];
    __shared__ float us[NVAC][H];
    int w = threadIdx.x >> 5;
    int lane = threadIdx.x & 31;
    float r = radius[0];
    float ang = 6.283185307179586f * (float)w / 16.0f;
    float vx = center[0] + __cosf(ang) * r;
    float vy = center[1] + __sinf(ang) * r;
    float vz = center[2] + (-0.5f + (float)w * (1.0f / 15.0f)) * r;

    for (int i = lane; i < H; i += 32) {
        float z0 = OMEGA * (W0[i] * vx + W0[H + i] * vy + W0[2 * H + i] * vz + b0[i]);
        h0s[w][i] = __sinf(z0);
        us[w][i]  = OMEGA * __cosf(z0);
    }
    __syncthreads();

    float psi = 0.f, g0 = 0.f, g1s = 0.f, g2 = 0.f, d0 = 0.f, d1 = 0.f, d2 = 0.f;
    const float o2 = OMEGA * OMEGA;
    for (int jj = 0; jj < 4; ++jj) {
        int j = lane + 32 * jj;
        float s1 = 0.f, a0 = 0.f, a1 = 0.f, a2 = 0.f, q0 = 0.f, q1 = 0.f, q2 = 0.f;
        for (int i = 0; i < H; ++i) {
            float w1 = W1[i * H + j];
            float hh = h0s[w][i], uu = us[w][i];
            float p0 = W0[i], p1 = W0[H + i], p2 = W0[2 * H + i];
            s1 += w1 * hh;
            a0 += w1 * uu * p0; a1 += w1 * uu * p1; a2 += w1 * uu * p2;
            q0 += w1 * hh * p0 * p0; q1 += w1 * hh * p1 * p1; q2 += w1 * hh * p2 * p2;
        }
        float z1 = OMEGA * (s1 + b1[j]);
        float h1 = __sinf(z1);
        float t  = OMEGA * __cosf(z1);
        float w2 = W2[j];
        psi += w2 * h1;
        g0 += w2 * t * a0; g1s += w2 * t * a1; g2 += w2 * t * a2;
        d0 += w2 * (-o2 * h1 * a0 * a0 - t * o2 * q0);
        d1 += w2 * (-o2 * h1 * a1 * a1 - t * o2 * q1);
        d2 += w2 * (-o2 * h1 * a2 * a2 - t * o2 * q2);
    }
    for (int msk = 1; msk < 32; msk <<= 1) {
        psi += __shfl_xor(psi, msk);
        g0 += __shfl_xor(g0, msk); g1s += __shfl_xor(g1s, msk); g2 += __shfl_xor(g2, msk);
        d0 += __shfl_xor(d0, msk); d1 += __shfl_xor(d1, msk); d2 += __shfl_xor(d2, msk);
    }
    if (lane == 0) {
        float psic = fminf(50.f, fmaxf(-50.f, nn0(psi + b2[0])));
        psiOut[w] = psic;
        float gr[3] = { g0, g1s, g2 };
        float dr[3] = { d0, d1, d2 };
        float lap = 0.f, gn2 = 0.f;
        for (int k = 0; k < 3; ++k) {
            float gc = clamp25(nn0(gr[k]));
            gvOut[w * 3 + k] = gc;
            gn2 += gc * gc;
            float gate = (__builtin_isfinite(gr[k]) && fabsf(gr[k]) < 25.0f) ? 1.0f : 0.0f;
            lap += gate * dr[k];
        }
        lap = clamp25(nn0(lap));
        specOut[w] = log1pf(psic * psic + 0.5f * gn2 + 0.25f * lap * lap);
    }
}

// ---------------------------------------------------------------------------
__global__ void alpha_kernel(const float* __restrict__ a, const float* __restrict__ at,
                             int n, float* __restrict__ out) {
    __shared__ float sm[256];
    int tid = threadIdx.x;
    float s = 0.f;
    for (int i = tid; i < n; i += 256) { float d = a[i] - at[i]; s += d * d; }
    sm[tid] = s;
    __syncthreads();
    for (int st = 128; st > 0; st >>= 1) {
        if (tid < st) sm[tid] += sm[tid + st];
        __syncthreads();
    }
    if (tid == 0) out[0] = sm[0] / (float)n;
}

__global__ void final_kernel(const float* __restrict__ hub, int nhub, long long N3,
                             const float* __restrict__ spec, const float* __restrict__ alpha,
                             float* __restrict__ out4) {
    __shared__ float sm[256];
    int tid = threadIdx.x;
    float s = 0.f;
    for (int i = tid; i < nhub; i += 256) s += hub[i];
    sm[tid] = s;
    __syncthreads();
    for (int st = 128; st > 0; st >>= 1) {
        if (tid < st) sm[tid] += sm[tid + st];
        __syncthreads();
    }
    if (tid == 0) {
        float gl = sm[0] / (float)N3;
        float sp = 0.f;
        for (int i = 0; i < 16; ++i) sp += spec[i];
        sp *= (1.0f / 16.0f);
        float al = alpha[0];
        out4[0] = gl; out4[1] = sp; out4[2] = al;
        out4[3] = gl + 0.05f * sp + 0.05f * al;
    }
}

// ---------------------------------------------------------------------------
extern "C" void kernel_launch(void* const* d_in, const int* in_sizes, int n_in,
                              void* d_out, int out_size, void* d_ws, size_t ws_size,
                              hipStream_t stream) {
    const float* pos     = (const float*)d_in[0];
    const float* forces  = (const float*)d_in[1];
    const float* alpha   = (const float*)d_in[2];
    const float* alpha_t = (const float*)d_in[3];
    const float* W0 = (const float*)d_in[4];
    const float* b0 = (const float*)d_in[5];
    const float* W1 = (const float*)d_in[6];
    const float* b1 = (const float*)d_in[7];
    const float* W2 = (const float*)d_in[8];
    const float* b2 = (const float*)d_in[9];
    int N = in_sizes[0] / 3;
    int nAlpha = in_sizes[2];

    float* out    = (float*)d_out;
    float* grads  = out + 4;
    float* psiOut = out + 4 + (size_t)N * 3;
    float* gvOut  = psiOut + 16;

    float* ws = (float*)d_ws;
    float* wcenter = ws + 0;     // 3
    float* wradius = ws + 3;     // 1
    float* walpha  = ws + 4;     // 1
    float* wspec   = ws + 8;     // 16
    float* wsums   = ws + 32;    // 256*3
    float* wmaxs   = wsums + 256 * 3;  // 256
    float* whub    = wmaxs + 256;      // nBlocks
    int nBlocks = (N + 63) / 64;
    _Float16* w1f = (_Float16*)(whub + nBlocks);
    _Float16* w1b = w1f + 16384;

    sum_pos_kernel<<<256, 256, 0, stream>>>(pos, N, wsums);
    finish_center_kernel<<<1, 32, 0, stream>>>(wsums, 256, N, wcenter);
    max_rad_kernel<<<256, 256, 0, stream>>>(pos, N, wcenter, wmaxs);
    finish_rad_kernel<<<1, 32, 0, stream>>>(wmaxs, 256, wradius);
    prep_w1_kernel<<<64, 256, 0, stream>>>(W1, w1f, w1b);
    field_grad_kernel<<<nBlocks, 128, 0, stream>>>(pos, forces, W0, b0, b1, W2,
                                                   w1f, w1b, wcenter, grads, whub, N);
    vacuum_kernel<<<1, 512, 0, stream>>>(W0, b0, W1, b1, W2, b2, wcenter, wradius,
                                         psiOut, gvOut, wspec);
    alpha_kernel<<<1, 256, 0, stream>>>(alpha, alpha_t, nAlpha, walpha);
    final_kernel<<<1, 256, 0, stream>>>(whub, nBlocks, (long long)N * 3, wspec, walpha, out);
}
